// SimpleHeteroConv_89163521065076
// MI455X (gfx1250) — matrix operand order repaired and checked by reading, not compiled, not run on hardware
//
#include <hip/hip_runtime.h>
#include <hip/hip_bf16.h>

typedef __attribute__((ext_vector_type(16))) __bf16 v16bf;
typedef __attribute__((ext_vector_type(8)))  float  v8f;

#define D_IN  128
#define D_OUT 128
#define NTYPES 2
// fragment table: [t][kt(4)][nt(8)][lane(32)][16 bf16]
#define FRAG_ELEMS (NTYPES * 4 * 8 * 32 * 16)

__host__ __device__ static inline unsigned bf16_bits(float f) {
    union { float f; unsigned u; } cv; cv.f = f;
    unsigned u = cv.u;
    // round-to-nearest-even f32 -> bf16
    return (u + 0x7FFFu + ((u >> 16) & 1u)) >> 16;
}

__host__ __device__ static inline __bf16 f2bf(float f) {
    unsigned short us = (unsigned short)bf16_bits(f);
    __bf16 b;
    __builtin_memcpy(&b, &us, 2);
    return b;
}

// Pack two floats into one dword holding {hi: bf16(hi), lo: bf16(lo)}.
__device__ static inline unsigned pk2bf(float lo, float hi) {
#if __has_builtin(__builtin_amdgcn_cvt_pk_bf16_f32)
    auto r = __builtin_amdgcn_cvt_pk_bf16_f32(lo, hi);   // dst.lo = lo, dst.hi = hi
    unsigned u;
    __builtin_memcpy(&u, &r, 4);
    return u;
#else
    return (bf16_bits(hi) << 16) | bf16_bits(lo);
#endif
}

// Guaranteed-hardware f32 atomic add (no return -> STOREcnt only;
// s_endpgm's implicit wait-idle covers completion before kernel end).
__device__ static inline void hw_atomic_add_f32(float* p, float v) {
    asm volatile("global_atomic_add_f32 %0, %1, off"
                 :
                 : "v"(p), "v"(v)
                 : "memory");
}

// Repack a [NTYPES, 128, 128] f32 weight into bf16 WMMA B-fragment order.
// One thread per (t, kt, nt, lane) -> 2048 threads; each writes 16 bf16 (32B).
__global__ void build_wfrag(const float* __restrict__ W, __bf16* __restrict__ frag) {
    int idx = blockIdx.x * blockDim.x + threadIdx.x;
    if (idx >= NTYPES * 4 * 8 * 32) return;
    int lane = idx & 31;
    int nt   = (idx >> 5) & 7;
    int kt   = (idx >> 8) & 3;
    int t    = (idx >> 10) & 1;
    int n  = lane & 15;
    int kb = (lane < 16) ? 0 : 8;    // B 32x16 bf16: lanes 0-15 hold K=0-7,16-23, lanes 16-31 K=8-15,24-31
    __bf16* o = frag + (size_t)idx * 16;
#pragma unroll
    for (int j = 0; j < 8; ++j) {
        int k0 = kt * 32 + kb + 2 * j + ((j >> 2) << 3);
        o[2 * j + 0] = f2bf(W[(size_t)(t * D_IN + k0 + 0) * D_OUT + nt * 16 + n]);
        o[2 * j + 1] = f2bf(W[(size_t)(t * D_IN + k0 + 1) * D_OUT + nt * 16 + n]);
    }
}

// Y[row,:] = X[row,:] @ W[ntype[row]]   (per 16-row tile; ntype is sorted so at
// most one tile is mixed -> that tile computes both types and selects per row)
__global__ void __launch_bounds__(128) typed_gemm(
        const float* __restrict__ X, const int* __restrict__ ntype,
        const __bf16* __restrict__ wfrag, float* __restrict__ Y, int N) {
    const int lane = threadIdx.x & 31;
    const int wave = threadIdx.x >> 5;
    const int tile = blockIdx.x * (blockDim.x >> 5) + wave;
    const int row0 = tile * 16;
    if (row0 >= N) return;

    const int half = lane >> 4;       // which 16-lane half
    const int m    = lane & 15;
    const int arow = row0 + m;        // A layout: both halves carry M=0..15
    const bool avalid = arow < N;

    // Build A fragments for all 4 K-steps.
    // ISA 16-bit A 16x32 layout per lane: two contiguous 8-value K runs:
    //   k = kt*32 + 8*half + [0..7]  -> elements 0..7
    //   k = kt*32 + 16 + 8*half + [0..7] -> elements 8..15
    // => 4 float4 loads per fragment, packed pairwise into dwords.
    v16bf afr[4];
    const float4* xr = avalid ? (const float4*)(X + (size_t)arow * D_IN) : nullptr;
#pragma unroll
    for (int kt = 0; kt < 4; ++kt) {
        float4 p0 = {0.f, 0.f, 0.f, 0.f}, p1 = p0, p2 = p0, p3 = p0;
        if (avalid) {
            int base = kt * 8 + half * 2;     // float4 index of k = kt*32 + 8*half
            p0 = xr[base + 0];
            p1 = xr[base + 1];
            p2 = xr[base + 4];                // k = kt*32 + 16 + 8*half
            p3 = xr[base + 5];
        }
        union { unsigned u[8]; v16bf v; } au;
        au.u[0] = pk2bf(p0.x, p0.y);
        au.u[1] = pk2bf(p0.z, p0.w);
        au.u[2] = pk2bf(p1.x, p1.y);
        au.u[3] = pk2bf(p1.z, p1.w);
        au.u[4] = pk2bf(p2.x, p2.y);
        au.u[5] = pk2bf(p2.z, p2.w);
        au.u[6] = pk2bf(p3.x, p3.y);
        au.u[7] = pk2bf(p3.z, p3.w);
        afr[kt] = au.v;
    }

    const int lastrow = (row0 + 15 < N) ? (row0 + 15) : (N - 1);
    const int t0 = ntype[row0];
    const int t1 = ntype[lastrow];
    const bool mixed = (t0 != t1);

    // For the (rare) mixed tile: which C rows use the second type.
    unsigned selmask = 0;
    if (mixed) {
#pragma unroll
        for (int r = 0; r < 8; ++r) {
            int rr = row0 + r + 8 * half;
            if (rr < N && ntype[rr] != t0) selmask |= (1u << r);
        }
    }

    const v16bf* wf = (const v16bf*)wfrag;

    for (int nt = 0; nt < 8; ++nt) {
        v8f acc0 = {0.f, 0.f, 0.f, 0.f, 0.f, 0.f, 0.f, 0.f};
#pragma unroll
        for (int kt = 0; kt < 4; ++kt) {
            v16bf b = wf[((size_t)(t0 * 4 + kt) * 8 + nt) * 32 + lane];
            acc0 = __builtin_amdgcn_wmma_f32_16x16x32_bf16(
                false, afr[kt], false, b, (short)0, acc0, false, false);
        }
        v8f acc1 = acc0;
        if (mixed) {
            v8f a1 = {0.f, 0.f, 0.f, 0.f, 0.f, 0.f, 0.f, 0.f};
#pragma unroll
            for (int kt = 0; kt < 4; ++kt) {
                v16bf b = wf[((size_t)(t1 * 4 + kt) * 8 + nt) * 32 + lane];
                a1 = __builtin_amdgcn_wmma_f32_16x16x32_bf16(
                    false, afr[kt], false, b, (short)0, a1, false, false);
            }
            acc1 = a1;
        }
        // C/D layout: VGPR r holds (M = r + 8*half, N = lane&15)
#pragma unroll
        for (int r = 0; r < 8; ++r) {
            int rr = row0 + r + 8 * half;
            if (rr < N) {
                float val = (mixed && ((selmask >> r) & 1u)) ? acc1[r] : acc0[r];
                Y[(size_t)rr * D_OUT + nt * 16 + (lane & 15)] = val;
            }
        }
    }
}

// h[dst[e], :] += v[src[e], :]  — one wave per edge, float4 per lane,
// 4 hardware global_atomic_add_f32 per lane into the L2-resident accumulator.
__global__ void __launch_bounds__(256) edge_scatter(
        const float* __restrict__ v, const int* __restrict__ src,
        const int* __restrict__ dst, float* __restrict__ h, int E) {
    int e = (int)((blockIdx.x * (size_t)blockDim.x + threadIdx.x) >> 5);
    int lane = threadIdx.x & 31;
    if (e >= E) return;
    int s = __builtin_amdgcn_readfirstlane(src[e]);
    int d = __builtin_amdgcn_readfirstlane(dst[e]);
    const float4* vp = (const float4*)(v + (size_t)s * D_IN);
    float4 val = vp[lane];                       // 32 lanes x 16B = 512B row
    float* hp = h + (size_t)d * D_IN + lane * 4;
    hw_atomic_add_f32(hp + 0, val.x);
    hw_atomic_add_f32(hp + 1, val.y);
    hw_atomic_add_f32(hp + 2, val.z);
    hw_atomic_add_f32(hp + 3, val.w);
}

extern "C" void kernel_launch(void* const* d_in, const int* in_sizes, int n_in,
                              void* d_out, int out_size, void* d_ws, size_t ws_size,
                              hipStream_t stream) {
    const float* x     = (const float*)d_in[0];
    const int*   ntype = (const int*)d_in[1];
    const int*   src   = (const int*)d_in[2];
    const int*   dst   = (const int*)d_in[3];
    const float* Wv    = (const float*)d_in[4];
    const float* Wa    = (const float*)d_in[5];
    float* out = (float*)d_out;

    const int N = in_sizes[0] / D_IN;   // 100000
    const int E = in_sizes[2];          // 800000

    char* ws = (char*)d_ws;
    size_t feat_bytes = (size_t)N * D_IN * sizeof(float);
    float*  v     = (float*)ws;
    float*  h     = (float*)(ws + feat_bytes);
    __bf16* fragV = (__bf16*)(ws + 2 * feat_bytes);
    __bf16* fragA = fragV + FRAG_ELEMS;

    // 1) repack weights into bf16 WMMA fragments
    build_wfrag<<<8, 256, 0, stream>>>(Wv, fragV);
    build_wfrag<<<8, 256, 0, stream>>>(Wa, fragA);

    // 2) zero the scatter accumulator
    hipMemsetAsync(h, 0, feat_bytes, stream);

    // 3) v = typed_linear(x, W_v, ntype)
    const int tiles = (N + 15) / 16;
    const int wpb = 4;                  // waves per block (128 threads)
    dim3 gblk((tiles + wpb - 1) / wpb);
    typed_gemm<<<gblk, wpb * 32, 0, stream>>>(x, ntype, fragV, v, N);

    // 4) h = segment_sum(v[src], dst)
    int edge_waves_per_block = 8;       // 256 threads
    dim3 eblk((E + edge_waves_per_block - 1) / edge_waves_per_block);
    edge_scatter<<<eblk, 256, 0, stream>>>(v, src, dst, h, E);

    // 5) out = typed_linear(h, W_a, ntype)
    typed_gemm<<<gblk, wpb * 32, 0, stream>>>(h, ntype, fragA, out, N);
}
